// MultiChannelAttention_62234076119700
// MI455X (gfx1250) — matrix operand order repaired and checked by reading, not compiled, not run on hardware
//
#include <hip/hip_runtime.h>
#include <hip/hip_bf16.h>

// ---------------------------------------------------------------------------
// MultiChannelAttention, MI455X (gfx1250), wave32 + v_wmma_f32_16x16x32_bf16
// bf16 pre-converted operands, n-major B, double-buffered async->LDS staging
// with partial s_wait_asynccnt so 3 loads stay in flight during WMMA compute.
//   M[b,h]    = wc[b] @ W[h]            (bf16 out)
//   P[b,h]    = M[b,h] @ ctx[b]         (f32 out)
//   attn      = softmax_rows(P)         (f32 in place + bf16 copy)
//   attn_out  = sum_h attn -> [B,L,IDF]
//   Z[b,h]    = attn[b,h] @ ctx[b]^T    (bf16 out, aliases M)
//   ctx_out[b]= sum_h Z[b,h] @ W[h]^T   (f32 out)
// ---------------------------------------------------------------------------

#define B_   32
#define IDF_ 64
#define CDF_ 256
#define SS_  1024
#define H_   8
#define L_   256

typedef __attribute__((ext_vector_type(16))) __bf16 v16bf;
typedef __attribute__((ext_vector_type(8)))  float  v8f;
typedef __attribute__((ext_vector_type(4)))  int    v4i;

#if __has_builtin(__builtin_amdgcn_global_load_async_to_lds_b128) && \
    __has_builtin(__builtin_amdgcn_s_wait_asynccnt)
#define HAVE_ASYNC 1
#else
#define HAVE_ASYNC 0
#endif

typedef __attribute__((address_space(1))) v4i gb_v4i;   // global int4
typedef __attribute__((address_space(3))) v4i ds_v4i;   // LDS int4

__device__ __forceinline__ unsigned short f2bf(float f) {
  union { __bf16 b; unsigned short s; } u;
  u.b = (__bf16)f;                       // hardware cvt path
  return u.s;
}

// 16-byte global -> LDS copy (async-to-LDS when the toolchain exposes it)
__device__ __forceinline__ void cp16(const unsigned short* g, unsigned short* l) {
#if HAVE_ASYNC
  __builtin_amdgcn_global_load_async_to_lds_b128(
      (gb_v4i*)(g), (ds_v4i*)(l), 0, 0);
#else
  *(uint4*)(void*)l = *(const uint4*)(const void*)g;
#endif
}

// ---------------------------------------------------------------------------
// GEMM core: C(64 x 128-block) += sum_it A_it(64 x K) @ B_it(K x 128)
// A: bf16 row-major [m][k] (lda). B: bf16 n-major [n][k] (ldb).
// 256 threads = 8 wave32 waves in 2(M) x 4(N); each wave: 32x32 = 4 WMMA accs.
// Double-buffered LDS; next stage's async loads overlap current WMMA work.
// ---------------------------------------------------------------------------
__device__ __forceinline__ void gemm_core(
    const unsigned short* __restrict__ A0, int lda, size_t aStride,
    const unsigned short* __restrict__ B0, int ldb, size_t bStride,
    int iters, int K, v8f acc[2][2]) {
  __shared__ alignas(16) unsigned short sA[2][64 * 32];
  __shared__ alignas(16) unsigned short sB[2][128 * 32];

  const int tid  = threadIdx.x;
  const int lane = tid & 31;
  const int wave = tid >> 5;
  const int mOff = (wave & 1) * 32;
  const int nOff = (wave >> 1) * 32;
  const int row  = lane & 15;
  const int kh   = lane >> 4;
  const int am   = tid >> 2;             // stage row index (A and B j=0)
  const int bn1  = (tid + 256) >> 2;     // B j=1 row index
  const int kq   = (tid & 3) * 8;        // 8-elem chunk within 32-wide k

  const int ksteps = K >> 5;
  const int S = iters * ksteps;

  auto stage = [&](int s, int bufsel) {
    int it = s / ksteps;
    int k0 = (s - it * ksteps) << 5;
    const unsigned short* A  = A0 + (size_t)it * aStride;
    const unsigned short* Bm = B0 + (size_t)it * bStride;
    cp16(A  + (size_t)am  * lda + k0 + kq, &sA[bufsel][am  * 32 + kq]);
    cp16(Bm + (size_t)am  * ldb + k0 + kq, &sB[bufsel][am  * 32 + kq]);
    cp16(Bm + (size_t)bn1 * ldb + k0 + kq, &sB[bufsel][bn1 * 32 + kq]);
  };

  stage(0, 0);
  for (int s = 0; s < S; ++s) {
    const int buf = s & 1;
    if (s + 1 < S) {
      stage(s + 1, buf ^ 1);             // 3 more async loads in flight
#if HAVE_ASYNC
      __builtin_amdgcn_s_wait_asynccnt(3);   // wait only for current stage
#endif
    } else {
#if HAVE_ASYNC
      __builtin_amdgcn_s_wait_asynccnt(0);
#endif
    }
    __syncthreads();

    // fragments straight from LDS (no conversion)
    union AB { v16bf v; uint4 q[2]; };
    AB a[2], b[2];
#pragma unroll
    for (int mt = 0; mt < 2; ++mt) {
      const unsigned short* ar = &sA[buf][(mOff + mt * 16 + row) * 32];
      a[mt].q[0] = *(const uint4*)(ar + kh * 8);        // e0..7  : k=kh*8+0..7
      a[mt].q[1] = *(const uint4*)(ar + 16 + kh * 8);   // e8..15 : k=16+kh*8..
    }
#pragma unroll
    for (int nt = 0; nt < 2; ++nt) {
      const unsigned short* br = &sB[buf][(nOff + nt * 16 + row) * 32];
      b[nt].q[0] = *(const uint4*)(br + kh * 8);        // k = kh*8 + 0..7
      b[nt].q[1] = *(const uint4*)(br + 16 + kh * 8);   // k = 16 + kh*8 + 0..7
    }
#pragma unroll
    for (int mt = 0; mt < 2; ++mt)
#pragma unroll
      for (int nt = 0; nt < 2; ++nt)
        acc[mt][nt] = __builtin_amdgcn_wmma_f32_16x16x32_bf16(
            false, a[mt].v, false, b[nt].v, (short)0, acc[mt][nt],
            false, false);
    __syncthreads();   // protect buf before stage(s+2) overwrites it
  }
}

__device__ __forceinline__ void zero_acc(v8f acc[2][2]) {
#pragma unroll
  for (int m = 0; m < 2; ++m)
#pragma unroll
    for (int n = 0; n < 2; ++n)
#pragma unroll
      for (int i = 0; i < 8; ++i) acc[m][n][i] = 0.f;
}

// D layout: VGPR r -> M = r + 8*(lane>=16), N = lane&15
template <bool TOBF16>
__device__ __forceinline__ void store_tile(void* Cv, int ldc,
                                           const v8f acc[2][2]) {
  const int tid  = threadIdx.x;
  const int lane = tid & 31;
  const int wave = tid >> 5;
  const int mOff = (wave & 1) * 32;
  const int nOff = (wave >> 1) * 32;
  const int kh   = lane >> 4;
  const int col  = lane & 15;
#pragma unroll
  for (int mt = 0; mt < 2; ++mt)
#pragma unroll
    for (int nt = 0; nt < 2; ++nt)
#pragma unroll
      for (int r = 0; r < 8; ++r) {
        size_t idx = (size_t)(mOff + mt * 16 + r + kh * 8) * ldc +
                     nOff + nt * 16 + col;
        if (TOBF16) ((unsigned short*)Cv)[idx] = f2bf(acc[mt][nt][r]);
        else        ((float*)Cv)[idx]          = acc[mt][nt][r];
      }
}

// ---------------- conversion / transpose passes ----------------
// f32 -> bf16 copy, 8 elements per thread (fixed 2,097,152-element arrays)
__global__ void mca_cvt(const float* __restrict__ in,
                        unsigned short* __restrict__ out) {
  int i = blockIdx.x * 256 + threadIdx.x;
  const float4* I = (const float4*)in;
  float4 x = I[i * 2], y = I[i * 2 + 1];
  union { uint4 q; unsigned short s[8]; } r;
  r.s[0] = f2bf(x.x); r.s[1] = f2bf(x.y); r.s[2] = f2bf(x.z); r.s[3] = f2bf(x.w);
  r.s[4] = f2bf(y.x); r.s[5] = f2bf(y.y); r.s[6] = f2bf(y.z); r.s[7] = f2bf(y.w);
  *(uint4*)(out + (size_t)i * 8) = r.q;
}

// [G][R][C] f32 -> [G][C][R] bf16, 32x32 LDS tiles
__global__ void mca_transpose_cvt(const float* __restrict__ in,
                                  unsigned short* __restrict__ out,
                                  int R, int C) {
  __shared__ float t[32][33];
  const float* I = in + (size_t)blockIdx.z * R * C;
  unsigned short* O = out + (size_t)blockIdx.z * R * C;
  int c0 = blockIdx.x * 32, r0 = blockIdx.y * 32;
  int tx = threadIdx.x, ty = threadIdx.y;   // 32 x 8
#pragma unroll
  for (int i = 0; i < 32; i += 8)
    t[ty + i][tx] = I[(size_t)(r0 + ty + i) * C + c0 + tx];
  __syncthreads();
#pragma unroll
  for (int i = 0; i < 32; i += 8)
    O[(size_t)(c0 + ty + i) * R + r0 + tx] = f2bf(t[tx][ty + i]);
}

// ---------------- GEMM wrappers ----------------
// K1: M[b,h](64x256,bf16) = wc[b](64x1024) @ W[h](1024x256); B = WT n-major
__global__ void mca_gemm_m(const unsigned short* __restrict__ wcB,
                           const unsigned short* __restrict__ WT,
                           unsigned short* __restrict__ M) {
  int nb = blockIdx.x & 1, bh = blockIdx.x >> 1, b = bh >> 3, h = bh & 7;
  v8f acc[2][2]; zero_acc(acc);
  gemm_core(wcB + (size_t)b * IDF_ * SS_, SS_, 0,
            WT + (size_t)h * CDF_ * SS_ + (size_t)(nb * 128) * SS_, SS_, 0,
            1, SS_, acc);
  store_tile<true>(M + (size_t)bh * IDF_ * CDF_ + nb * 128, CDF_, acc);
}

// K2: P[b,h](64x256,f32) = M[b,h] @ ctx[b]; B = ctxT n-major
__global__ void mca_gemm_attn(const unsigned short* __restrict__ M,
                              const unsigned short* __restrict__ ctxT,
                              float* __restrict__ P) {
  int nb = blockIdx.x & 1, bh = blockIdx.x >> 1, b = bh >> 3;
  v8f acc[2][2]; zero_acc(acc);
  gemm_core(M + (size_t)bh * IDF_ * CDF_, CDF_, 0,
            ctxT + (size_t)b * L_ * CDF_ + (size_t)(nb * 128) * CDF_, CDF_, 0,
            1, CDF_, acc);
  store_tile<false>(P + (size_t)bh * IDF_ * L_ + nb * 128, L_, acc);
}

// K5: Z[b,h](64x256,bf16) = attn[b,h] @ ctx[b]^T; B = ctxB (naturally n-major)
__global__ void mca_gemm_z(const unsigned short* __restrict__ attnB,
                           const unsigned short* __restrict__ ctxB,
                           unsigned short* __restrict__ Z) {
  int nb = blockIdx.x & 1, bh = blockIdx.x >> 1, b = bh >> 3;
  v8f acc[2][2]; zero_acc(acc);
  gemm_core(attnB + (size_t)bh * IDF_ * L_, L_, 0,
            ctxB + (size_t)b * CDF_ * L_ + (size_t)(nb * 128) * L_, L_, 0,
            1, L_, acc);
  store_tile<true>(Z + (size_t)bh * IDF_ * CDF_ + nb * 128, CDF_, acc);
}

// K6: ctx_out[b](64x1024,f32) = sum_h Z[b,h] @ W[h]^T; B = Wb n-major
__global__ void mca_gemm_out(const unsigned short* __restrict__ Z,
                             const unsigned short* __restrict__ Wb,
                             float* __restrict__ out) {
  int nb = blockIdx.x & 7, b = blockIdx.x >> 3;
  v8f acc[2][2]; zero_acc(acc);
  gemm_core(Z + (size_t)b * H_ * IDF_ * CDF_, CDF_, (size_t)IDF_ * CDF_,
            Wb + (size_t)(nb * 128) * CDF_, CDF_, (size_t)SS_ * CDF_,
            H_, CDF_, acc);
  store_tile<false>(out + (size_t)b * IDF_ * SS_ + nb * 128, SS_, acc);
}

// ---------------- softmax + head-sum ----------------
__global__ void mca_softmax(float* __restrict__ P,
                            unsigned short* __restrict__ attnB) {
  int row  = blockIdx.x * 8 + (threadIdx.x >> 5);
  int lane = threadIdx.x & 31;
  float* p = P + (size_t)row * L_;
  float v[8];
  float mx = -3.402823e38f;
#pragma unroll
  for (int i = 0; i < 8; ++i) { v[i] = p[lane + i * 32]; mx = fmaxf(mx, v[i]); }
#pragma unroll
  for (int o = 16; o > 0; o >>= 1) mx = fmaxf(mx, __shfl_xor(mx, o, 32));
  float s = 0.f;
#pragma unroll
  for (int i = 0; i < 8; ++i) { v[i] = __expf(v[i] - mx); s += v[i]; }
#pragma unroll
  for (int o = 16; o > 0; o >>= 1) s += __shfl_xor(s, o, 32);
  float inv = 1.0f / s;
#pragma unroll
  for (int i = 0; i < 8; ++i) {
    float r = v[i] * inv;
    p[lane + i * 32] = r;
    attnB[(size_t)row * L_ + lane + i * 32] = f2bf(r);
  }
}

// attn[B,H,IDF,L] -> out[B,L,IDF] summed over H. Coalesced both directions
// via a 64(i) x 32(l) LDS tile. grid = B*8 blocks of 256.
__global__ void mca_attnout(const float* __restrict__ attn,
                            float* __restrict__ out) {
  __shared__ float s[64][33];
  int b  = blockIdx.x >> 3;
  int lb = (blockIdx.x & 7) * 32;
  int tx = threadIdx.x & 31;             // l within tile (coalesced reads)
  int ty = threadIdx.x >> 5;             // i stride 8
#pragma unroll
  for (int i = ty; i < 64; i += 8) {
    float a = 0.f;
#pragma unroll
    for (int h = 0; h < H_; ++h)
      a += attn[(((size_t)(b * H_ + h) * IDF_) + i) * L_ + lb + tx];
    s[i][tx] = a;
  }
  __syncthreads();
#pragma unroll
  for (int l = ty; l < 32; l += 8) {     // coalesced writes over i
    out[((size_t)b * L_ + lb + l) * IDF_ + tx]      = s[tx][l];
    out[((size_t)b * L_ + lb + l) * IDF_ + tx + 32] = s[tx + 32][l];
  }
}

// ---------------------------------------------------------------------------
extern "C" void kernel_launch(void* const* d_in, const int* in_sizes, int n_in,
                              void* d_out, int out_size, void* d_ws, size_t ws_size,
                              hipStream_t stream) {
  const float* wc  = (const float*)d_in[0];  // [B, IDF, SS]
  const float* ctx = (const float*)d_in[1];  // [B, CDF, L]
  const float* W   = (const float*)d_in[2];  // [H, SS, CDF]
  float* out = (float*)d_out;                // ctx_out ++ attn_out

  const size_t N2M = (size_t)2097152;        // wc / ctx / W element count
  const size_t N4M = (size_t)4194304;        // M / P / attn / Z element count

  unsigned short* wcB  = (unsigned short*)d_ws;   // bf16 wc        (4MB)
  unsigned short* ctxB = wcB  + N2M;              // bf16 ctx       (4MB)
  unsigned short* ctxT = ctxB + N2M;              // bf16 ctx^T     (4MB)
  unsigned short* Wb   = ctxT + N2M;              // bf16 W         (4MB)
  unsigned short* WT   = Wb   + N2M;              // bf16 W^T       (4MB)
  unsigned short* MZ   = WT   + N2M;              // bf16 M, then Z (8MB)
  unsigned short* attnB= MZ   + N4M;              // bf16 attn      (8MB)
  float*          Pf   = (float*)(attnB + N4M);   // f32 P/attn     (16MB)

  mca_cvt<<<1024, 256, 0, stream>>>(wc,  wcB);
  mca_cvt<<<1024, 256, 0, stream>>>(ctx, ctxB);
  mca_cvt<<<1024, 256, 0, stream>>>(W,   Wb);
  mca_transpose_cvt<<<dim3(8, 8, 32),  dim3(32, 8), 0, stream>>>(ctx, ctxT, CDF_, L_);
  mca_transpose_cvt<<<dim3(8, 32, 8),  dim3(32, 8), 0, stream>>>(W,   WT,   SS_,  CDF_);

  mca_gemm_m   <<<B_ * H_ * 2, 256, 0, stream>>>(wcB, WT, MZ);
  mca_gemm_attn<<<B_ * H_ * 2, 256, 0, stream>>>(MZ, ctxT, Pf);
  mca_softmax  <<<(B_ * H_ * IDF_) / 8, 256, 0, stream>>>(Pf, attnB);
  mca_attnout  <<<B_ * 8, 256, 0, stream>>>(Pf, out + (size_t)B_ * IDF_ * SS_);
  mca_gemm_z   <<<B_ * H_ * 2, 256, 0, stream>>>(attnB, ctxB, MZ);
  mca_gemm_out <<<B_ * 8, 256, 0, stream>>>(MZ, Wb, out);
}
